// Net_27865747816548
// MI455X (gfx1250) — matrix operand order repaired and checked by reading, not compiled, not run on hardware
//
#include <hip/hip_runtime.h>

#define NN 20000      // nodes      (multiple of 32)
#define NE 640000     // edges
#define DIM 128
#define NG 128        // graphs     (multiple of 32)
#define NL 5          // layers
#define OC 10         // out channels
#define BN_EPS 1e-5f

typedef __attribute__((ext_vector_type(16))) __bf16 v16bf;
typedef __attribute__((ext_vector_type(8)))  __bf16 v8bf;
typedef __attribute__((ext_vector_type(4)))  __bf16 v4bf;
typedef __attribute__((ext_vector_type(8)))  float  v8f;

// ---------------------------------------------------------------------------
// Pack W[K=128][N=128] (row-major) into a WMMA B-operand lane layout (contiguous k, same order as the A loads):
// p[((t*4+kk)*32+lane)*16 + i] = bf16( W[kk*32 + (lane>>4)*16 + i][t*16 + (lane&15)] )
// -> per lane, per K-step: 16 contiguous bf16 (32B), fully coalesced loads.
// 11 matrices: 5x W1, 5x W2, 1x lin1_W.
// ---------------------------------------------------------------------------
__global__ __launch_bounds__(256) void pack_weights_kernel(
    const float* __restrict__ W1s, const float* __restrict__ W2s,
    const float* __restrict__ lin1W, __bf16* __restrict__ wp)
{
    int gid = blockIdx.x * 256 + threadIdx.x;
    if (gid >= 11 * 16384) return;
    int mat  = gid >> 14;
    int idx  = gid & 16383;
    int i    = idx & 15;
    int lane = (idx >> 4) & 31;
    int kk   = (idx >> 9) & 3;
    int t    = idx >> 11;
    int k = kk * 32 + (lane >> 4) * 16 + i;
    int n = t * 16 + (lane & 15);
    const float* src;
    if (mat < 5)       src = W1s + mat * 16384;
    else if (mat < 10) src = W2s + (mat - 5) * 16384;
    else               src = lin1W;
    wp[gid] = (__bf16)src[k * DIM + n];
}

// Per-column scale/shift for ALL 11 GEMMs (no branches in the GEMM epilogue):
//   slots 0..4  : GEMM1 of layer l : s = g*rsqrt(v+eps), sh = beta + s*(b1-mean)
//   slots 5..9  : GEMM2 of layer l : s = 1,              sh = b2
//   slot  10    : lin1             : s = 1,              sh = lin1_b
__global__ __launch_bounds__(256) void epilogue_precompute_kernel(
    const float* __restrict__ g, const float* __restrict__ bt,
    const float* __restrict__ mn, const float* __restrict__ vr,
    const float* __restrict__ b1, const float* __restrict__ b2,
    const float* __restrict__ lin1b,
    float* __restrict__ scale, float* __restrict__ shift)
{
    int i = blockIdx.x * 256 + threadIdx.x;
    if (i >= 11 * DIM) return;
    float s, sh;
    if (i < 5 * DIM) {
        s  = g[i] * rsqrtf(vr[i] + BN_EPS);
        sh = bt[i] + s * (b1[i] - mn[i]);
    } else if (i < 10 * DIM) {
        s  = 1.0f;
        sh = b2[i - 5 * DIM];
    } else {
        s  = 1.0f;
        sh = lin1b[i - 10 * DIM];
    }
    scale[i] = s;
    shift[i] = sh;
}

__global__ __launch_bounds__(256) void copy_kernel(
    const float* __restrict__ src, float* __restrict__ dst, int n4)
{
    int i = blockIdx.x * 256 + threadIdx.x;
    if (i < n4) ((float4*)dst)[i] = ((const float4*)src)[i];
}

__global__ __launch_bounds__(256) void zero_kernel(float* __restrict__ dst, int n4)
{
    int i = blockIdx.x * 256 + threadIdx.x;
    if (i < n4) ((float4*)dst)[i] = make_float4(0.f, 0.f, 0.f, 0.f);
}

// One wave per edge: coalesced 512B row gather, 4 f32 atomics per lane.
// Whole feature array is L2-resident (10.2 MB << 192 MB L2) -> L2 atomic bound.
__global__ __launch_bounds__(256) void edge_scatter_kernel(
    const float* __restrict__ feat, const int* __restrict__ ei,
    float* __restrict__ agg)
{
    unsigned gid = blockIdx.x * 256u + threadIdx.x;
    unsigned e = gid >> 5;
    if (e >= NE) return;
    int lane = gid & 31;
    int src = ei[e];
    int dst = ei[NE + e];
    float4 v = ((const float4*)(feat + (size_t)src * DIM))[lane];
    float* d = agg + (size_t)dst * DIM + lane * 4;
    atomicAdd(d + 0, v.x);
    atomicAdd(d + 1, v.y);
    atomicAdd(d + 2, v.z);
    atomicAdd(d + 3, v.w);
}

// GEMM: out[M x 128] = relu( (A[M x 128] @ W) * scale[n] + shift[n] ), M % 32 == 0.
// Block = 256 threads = 8 waves; block owns a 32-row stripe; each wave owns one
// 16-column tile with TWO accumulators (rows m0..m0+15 and m0+16..m0+31), so each
// B fragment loaded from global feeds two v_wmma_f32_16x16x32_bf16 ops.
// 8 WMMAs per wave, no bounds checks, straight-line epilogue.
__global__ __launch_bounds__(256) void gemm_bn_relu_kernel(
    const float* __restrict__ A, const __bf16* __restrict__ Wp,
    const float* __restrict__ scale, const float* __restrict__ shift,
    float* __restrict__ out)
{
    __shared__ __bf16 aT[32 * DIM];  // 8 KB: 32-row A stripe in bf16
    int tid = threadIdx.x;
    int m0 = blockIdx.x * 32;

    // Stage A: 32 x 128 f32 -> bf16 LDS (1024 float4, 4 per thread)
    #pragma unroll
    for (int it = 0; it < 4; ++it) {
        int v = tid + it * 256;
        int row = v >> 5;
        int c4 = v & 31;
        float4 f = ((const float4*)(A + (size_t)(m0 + row) * DIM))[c4];
        v4bf b4 = { (__bf16)f.x, (__bf16)f.y, (__bf16)f.z, (__bf16)f.w };
        *(v4bf*)(&aT[row * DIM + c4 * 4]) = b4;
    }
    __syncthreads();

    int wave = tid >> 5;       // N-tile index 0..7
    int lane = tid & 31;
    int hi = lane >> 4;
    int nl = lane & 15;

    v8f acc0 = {};
    v8f acc1 = {};
    #pragma unroll
    for (int kk = 0; kk < 4; ++kk) {
        // A fragments (contiguous, matches packed B): elems 0..15 -> K=kk*32+hi*16..
        union { v16bf v; v8bf h[2]; } a0, a1;
        const __bf16* ar0 = &aT[nl * DIM + kk * 32 + hi * 16];
        const __bf16* ar1 = &aT[(16 + nl) * DIM + kk * 32 + hi * 16];
        a0.h[0] = *(const v8bf*)(ar0);
        a0.h[1] = *(const v8bf*)(ar0 + 8);
        a1.h[0] = *(const v8bf*)(ar1);
        a1.h[1] = *(const v8bf*)(ar1 + 8);
        // B fragment: pre-packed, 32B contiguous per lane (coalesced), reused x2
        v16bf bv = *(const v16bf*)(Wp + (((wave * 4 + kk) * 32 + lane) << 4));
        acc0 = __builtin_amdgcn_wmma_f32_16x16x32_bf16(
            false, a0.v, false, bv, (short)0, acc0, false, false);
        acc1 = __builtin_amdgcn_wmma_f32_16x16x32_bf16(
            false, a1.v, false, bv, (short)0, acc1, false, false);
    }

    int n = wave * 16 + nl;
    float s  = scale[n];
    float sh = shift[n];
    #pragma unroll
    for (int r = 0; r < 8; ++r) {
        int m = m0 + hi * 8 + r;          // D layout: VGPR r -> row r + hi*8
        float v0 = acc0[r] * s + sh;
        float v1 = acc1[r] * s + sh;
        out[(size_t)m * DIM + n]        = v0 > 0.f ? v0 : 0.f;
        out[(size_t)(m + 16) * DIM + n] = v1 > 0.f ? v1 : 0.f;
    }
}

// One wave per node: pooled[batch[i]] += h[i]
__global__ __launch_bounds__(256) void pool_kernel(
    const float* __restrict__ h, const int* __restrict__ batch,
    float* __restrict__ pooled)
{
    unsigned gid = blockIdx.x * 256u + threadIdx.x;
    unsigned node = gid >> 5;
    if (node >= NN) return;
    int lane = gid & 31;
    int g = batch[node];
    float4 v = ((const float4*)(h + (size_t)node * DIM))[lane];
    float* d = pooled + (size_t)g * DIM + lane * 4;
    atomicAdd(d + 0, v.x);
    atomicAdd(d + 1, v.y);
    atomicAdd(d + 2, v.z);
    atomicAdd(d + 3, v.w);
}

// Final 128->10 projection + log_softmax; one wave per graph.
__global__ __launch_bounds__(32) void head_kernel(
    const float* __restrict__ h3, const float* __restrict__ W2,
    const float* __restrict__ b2, float* __restrict__ out)
{
    __shared__ float lg[OC];
    int g = blockIdx.x;
    int o = threadIdx.x;
    if (o < OC) {
        float acc = b2[o];
        for (int k = 0; k < DIM; ++k)
            acc += h3[g * DIM + k] * W2[k * OC + o];
        lg[o] = acc;
    }
    __syncthreads();
    if (o < OC) {
        float mx = -3.4e38f;
        for (int j = 0; j < OC; ++j) mx = fmaxf(mx, lg[j]);
        float s = 0.f;
        for (int j = 0; j < OC; ++j) s += __expf(lg[j] - mx);
        out[g * OC + o] = lg[o] - mx - __logf(s);
    }
}

extern "C" void kernel_launch(void* const* d_in, const int* in_sizes, int n_in,
                              void* d_out, int out_size, void* d_ws, size_t ws_size,
                              hipStream_t stream)
{
    const float* x      = (const float*)d_in[0];
    const int*   ei     = (const int*)d_in[1];
    const int*   batch  = (const int*)d_in[2];
    const float* W1s    = (const float*)d_in[3];
    const float* b1s    = (const float*)d_in[4];
    const float* gammas = (const float*)d_in[5];
    const float* betas  = (const float*)d_in[6];
    const float* means  = (const float*)d_in[7];
    const float* vars_  = (const float*)d_in[8];
    const float* W2s    = (const float*)d_in[9];
    const float* b2s    = (const float*)d_in[10];
    const float* lin1W  = (const float*)d_in[11];
    const float* lin1b  = (const float*)d_in[12];
    const float* lin2W  = (const float*)d_in[13];
    const float* lin2b  = (const float*)d_in[14];
    float* out = (float*)d_out;

    char* ws = (char*)d_ws;
    float*  agg    = (float*)(ws + 0);          // 10,240,000 B
    float*  hA     = (float*)(ws + 10240000);   // 10,240,000 B (current features)
    float*  hB     = (float*)(ws + 20480000);   // 10,240,000 B (GEMM1 out)
    __bf16* wp     = (__bf16*)(ws + 30720000);  // 11 * 32 KB packed bf16 weights
    float*  scale  = (float*)(ws + 31080448);   // 11*128 f32
    float*  shift  = (float*)(ws + 31086080);   // 11*128 f32
    float*  pooled = (float*)(ws + 31091712);   // 128*128 f32
    float*  h3     = (float*)(ws + 31157248);   // 128*128 f32

    pack_weights_kernel<<<704, 256, 0, stream>>>(W1s, W2s, lin1W, wp);
    epilogue_precompute_kernel<<<6, 256, 0, stream>>>(
        gammas, betas, means, vars_, b1s, b2s, lin1b, scale, shift);

    const float* cur = x;
    for (int l = 0; l < NL; ++l) {
        // agg = cur (self term, eps=0), then agg += sum_{j->i} cur[j]
        copy_kernel<<<2500, 256, 0, stream>>>(cur, agg, NN * DIM / 4);
        edge_scatter_kernel<<<NE * 32 / 256, 256, 0, stream>>>(cur, ei, agg);
        // h = relu(bn(agg @ W1 + b1))  (BN folded into scale/shift)
        gemm_bn_relu_kernel<<<NN / 32, 256, 0, stream>>>(
            agg, wp + l * 16384, scale + l * DIM, shift + l * DIM, hB);
        // h = relu(h @ W2 + b2)
        gemm_bn_relu_kernel<<<NN / 32, 256, 0, stream>>>(
            hB, wp + (5 + l) * 16384, scale + (5 + l) * DIM, shift + (5 + l) * DIM, hA);
        cur = hA;
    }

    zero_kernel<<<16, 256, 0, stream>>>(pooled, NG * DIM / 4);
    pool_kernel<<<NN * 32 / 256, 256, 0, stream>>>(hA, batch, pooled);
    // relu(pooled @ lin1_W + lin1_b) via the same WMMA GEMM (M=128 -> 4 blocks)
    gemm_bn_relu_kernel<<<NG / 32, 256, 0, stream>>>(
        pooled, wp + 10 * 16384, scale + 10 * DIM, shift + 10 * DIM, h3);
    head_kernel<<<NG, 32, 0, stream>>>(h3, lin2W, lin2b, out);
}
